// CausalSelfAttention_30880814858916
// MI455X (gfx1250) — matrix operand order repaired and checked by reading, compiled, not run on hardware
//
#include <hip/hip_runtime.h>
#include <hip/hip_bf16.h>

#define B_ 4
#define T_ 2048
#define C_ 1024
#define H_ 16
#define D_ 64

typedef __bf16 bf16;
typedef __attribute__((ext_vector_type(16))) __bf16 v16bf;
typedef __attribute__((ext_vector_type(8)))  __bf16 v8bf;
typedef __attribute__((ext_vector_type(8)))  float  v8f;

static __device__ __forceinline__ v8bf ld8(const bf16* p) { return *(const v8bf*)p; }

static __device__ __forceinline__ v16bf cat(v8bf lo, v8bf hi) {
    v16bf r;
#pragma unroll
    for (int i = 0; i < 8; ++i) { r[i] = lo[i]; r[i + 8] = hi[i]; }
    return r;
}

static __device__ __forceinline__ v8f wmma_bf16(v16bf a, v16bf b, v8f c) {
    return __builtin_amdgcn_wmma_f32_16x16x32_bf16(false, a, false, b, (short)0, c, false, false);
}

// ---------------- conversion kernels ----------------

__global__ void cvt_kernel(const float* __restrict__ src, unsigned short* __restrict__ dstu, int n) {
    int i = blockIdx.x * blockDim.x + threadIdx.x;
    if (i < n) ((bf16*)dstu)[i] = (bf16)src[i];
}

// dst[n*rows + k] = src[k*cols + n]   (weights [K][N] f32 -> [N][K] bf16)
__global__ void cvt_tr_kernel(const float* __restrict__ src, unsigned short* __restrict__ dstu,
                              int rows, int cols) {
    int i = blockIdx.x * blockDim.x + threadIdx.x;
    if (i < rows * cols) {
        int n = i / rows;
        int k = i - n * rows;
        ((bf16*)dstu)[i] = (bf16)src[(size_t)k * cols + n];
    }
}

// ---------------- fragment loaders (64x64 tile per wave) ----------------

static __device__ __forceinline__ void load_a4(const bf16* arow, int kb, int hlf, v16bf a[4]) {
    // arow = &A[mbase + l16][0], row stride C_; mt selects 16-row subtile
#pragma unroll
    for (int mt = 0; mt < 4; ++mt) {
        const bf16* p = arow + (size_t)mt * 16 * C_ + kb;
        a[mt] = cat(ld8(p + hlf * 8), ld8(p + 16 + hlf * 8));
    }
}

static __device__ __forceinline__ void load_b4(const bf16* brow, int kb, int hlf, v16bf b[4]) {
    // brow = &Wt[nbase + l16][0], row stride C_; nt selects 16-col subtile
#pragma unroll
    for (int nt = 0; nt < 4; ++nt) {
        const bf16* p = brow + (size_t)nt * 16 * C_ + kb;
        b[nt] = cat(ld8(p + hlf * 8), ld8(p + 16 + hlf * 8));
    }
}

#define WMMA_TILE16(A, Bv, ACC)                                          \
    _Pragma("unroll") for (int mt = 0; mt < 4; ++mt)                     \
    _Pragma("unroll") for (int nt = 0; nt < 4; ++nt)                     \
        ACC[mt][nt] = wmma_bf16(A[mt], Bv[nt], ACC[mt][nt]);

// ---------------- QKV GEMM: [8192,1024] x [1024,3072], 64x64/wave, double-buffered ----------------

__global__ __launch_bounds__(32) void gemm_qkv_kernel(
    const unsigned short* __restrict__ xbu, const unsigned short* __restrict__ wTu,
    const float* __restrict__ bias,
    unsigned short* __restrict__ qu, unsigned short* __restrict__ ku, unsigned short* __restrict__ vtu) {
    const bf16* xb = (const bf16*)xbu;
    const bf16* wT = (const bf16*)wTu;
    bf16* qws = (bf16*)qu; bf16* kws = (bf16*)ku; bf16* vt = (bf16*)vtu;

    const int lane = threadIdx.x & 31;
    const int hlf = lane >> 4, l16 = lane & 15;
    const int NB = (3 * C_) / 64;                          // 48
    int nb = blockIdx.x % NB, mb = blockIdx.x / NB;
    int mbase = mb * 64, nbase = nb * 64;

    v8f acc[4][4];
    v8f zero = {0.f, 0.f, 0.f, 0.f, 0.f, 0.f, 0.f, 0.f};
#pragma unroll
    for (int mt = 0; mt < 4; ++mt)
#pragma unroll
        for (int nt = 0; nt < 4; ++nt) acc[mt][nt] = zero;

    const bf16* arow = xb + (size_t)(mbase + l16) * C_;
    const bf16* brow = wT + (size_t)(nbase + l16) * C_;

    v16bf a0[4], b0[4], a1[4], b1[4];
    load_a4(arow, 0, hlf, a0);
    load_b4(brow, 0, hlf, b0);
    for (int kb = 0; kb < C_; kb += 64) {
        load_a4(arow, kb + 32, hlf, a1);
        load_b4(brow, kb + 32, hlf, b1);
        WMMA_TILE16(a0, b0, acc);
        if (kb + 64 < C_) {
            load_a4(arow, kb + 64, hlf, a0);
            load_b4(brow, kb + 64, hlf, b0);
        }
        WMMA_TILE16(a1, b1, acc);
    }

#pragma unroll
    for (int nt = 0; nt < 4; ++nt) {
        int n = nbase + nt * 16 + l16;
        int sec = n >> 10;                 // 0=q 1=k 2=v (uniform per nt-subtile)
        int c = n & (C_ - 1);
        int hh = c >> 6, d = c & 63;
        float bv = bias[n];
#pragma unroll
        for (int mt = 0; mt < 4; ++mt) {
#pragma unroll
            for (int i = 0; i < 8; ++i) {
                int row = mbase + mt * 16 + i + 8 * hlf;
                int bidx = row >> 11, t = row & (T_ - 1);
                bf16 val = (bf16)(acc[mt][nt][i] + bv);
                size_t hb = (size_t)(bidx * H_ + hh);
                if (sec == 0)      qws[(hb * T_ + t) * D_ + d] = val;
                else if (sec == 1) kws[(hb * T_ + t) * D_ + d] = val;
                else               vt[(hb * D_ + d) * T_ + t]  = val;
            }
        }
    }
}

// ---------------- flash attention: 4 waves / block, 64 q-rows, K/V shared via LDS ----------------

__global__ __launch_bounds__(128) void attn_kernel(
    const unsigned short* __restrict__ qu, const unsigned short* __restrict__ ku,
    const unsigned short* __restrict__ vtu, unsigned short* __restrict__ yu) {
    const bf16* qws = (const bf16*)qu;
    const bf16* kws = (const bf16*)ku;
    const bf16* vh0 = (const bf16*)vtu;
    bf16* yws = (bf16*)yu;

    const int tid = threadIdx.x;
    const int lane = tid & 31;
    const int wid = tid >> 5;                              // 0..3
    const int hlf = lane >> 4, l16 = lane & 15;
    const int QB = T_ / 64;                                // 32
    int qblk = blockIdx.x % QB;
    int h    = (blockIdx.x / QB) % H_;
    int b    = blockIdx.x / (QB * H_);
    const int qbase = qblk * 64;
    const int qw = qbase + wid * 16;                       // this wave's 16 q-rows
    const size_t hb = (size_t)(b * H_ + h);
    const bf16* qh = qws + hb * T_ * D_;
    const bf16* kh = kws + hb * T_ * D_;
    const bf16* vh = vh0 + hb * D_ * T_;

    // Q A-fragments (16 x 32 each, d=0..31 and d=32..63)
    const bf16* qr = qh + (size_t)(qw + l16) * D_;
    v16bf aQ0 = cat(ld8(qr + hlf * 16),      ld8(qr + hlf * 16 + 8));
    v16bf aQ1 = cat(ld8(qr + 32 + hlf * 16), ld8(qr + 40 + hlf * 16));

    float mrun[8], lrun[8];
    v8f yacc[4];
    v8f zero = {0.f, 0.f, 0.f, 0.f, 0.f, 0.f, 0.f, 0.f};
#pragma unroll
    for (int i = 0; i < 8; ++i) { mrun[i] = -1e30f; lrun[i] = 0.f; }
#pragma unroll
    for (int t = 0; t < 4; ++t) yacc[t] = zero;

    __shared__ __align__(16) bf16 Klds[32][72];            // 32 keys x 64 d (144B stride)
    __shared__ __align__(16) bf16 Vlds[64][40];            // 64 d x 32 keys (80B stride)
    __shared__ __align__(16) bf16 Plds[4][16][40];         // per-wave P staging

    const int nkt = qbase / 32 + 2;                        // 32-key blocks for the 64-row q block
    for (int kt = 0; kt < nkt; ++kt) {
        int kb = kt * 32;
        __syncthreads();                                   // prior iter's LDS reads complete
#pragma unroll
        for (int j = 0; j < 2; ++j) {                      // K tile: 32x64 bf16 = 256 x 16B chunks
            int c = tid + j * 128;
            int row = c >> 3, e8 = (c & 7) * 8;
            *(v8bf*)&Klds[row][e8] = ld8(kh + (size_t)(kb + row) * D_ + e8);
        }
#pragma unroll
        for (int j = 0; j < 2; ++j) {                      // V^T tile: 64x32 bf16 = 256 x 16B chunks
            int c = tid + j * 128;
            int row = c >> 2, e8 = (c & 3) * 8;
            *(v8bf*)&Vlds[row][e8] = ld8(vh + (size_t)row * T_ + kb + e8);
        }
        // prefetch next tile into cache while this one is consumed (global_prefetch_b8)
        if (kt + 1 < nkt) {
            int kb2 = kb + 32;
            if (tid < 32) __builtin_prefetch(kh + (size_t)(kb2 + tid) * D_, 0, 3);   // 32 x 128B lines
            if (tid < 64) __builtin_prefetch(vh + (size_t)tid * T_ + kb2, 0, 3);     // 64 rows x 64B
        }
        __syncthreads();

        // S = Q.K^T for this wave's 16 rows x 32 keys
        v8f s0 = zero, s1 = zero;
        {
            int r = l16;
            v16bf kb0 = cat(ld8(&Klds[r][hlf * 16]),      ld8(&Klds[r][hlf * 16 + 8]));
            v16bf kb1 = cat(ld8(&Klds[r][32 + hlf * 16]), ld8(&Klds[r][32 + hlf * 16 + 8]));
            s0 = wmma_bf16(aQ0, kb0, s0);
            s0 = wmma_bf16(aQ1, kb1, s0);
        }
        {
            int r = 16 + l16;
            v16bf kb0 = cat(ld8(&Klds[r][hlf * 16]),      ld8(&Klds[r][hlf * 16 + 8]));
            v16bf kb1 = cat(ld8(&Klds[r][32 + hlf * 16]), ld8(&Klds[r][32 + hlf * 16 + 8]));
            s1 = wmma_bf16(aQ0, kb0, s1);
            s1 = wmma_bf16(aQ1, kb1, s1);
        }

        // online softmax in C-layout: row = qw + i + 8*hlf, col = kb + sub*16 + l16
#pragma unroll
        for (int i = 0; i < 8; ++i) {
            int row = qw + i + 8 * hlf;
            float e0 = ((kb + l16) <= row)      ? s0[i] * 0.125f : -1e30f;
            float e1 = ((kb + 16 + l16) <= row) ? s1[i] * 0.125f : -1e30f;
            float m = fmaxf(e0, e1);
#pragma unroll
            for (int off = 1; off < 16; off <<= 1) m = fmaxf(m, __shfl_xor(m, off, 32));
            float mnew = fmaxf(mrun[i], m);
            float alpha = __expf(mrun[i] - mnew);
            float p0 = __expf(e0 - mnew);
            float p1 = __expf(e1 - mnew);
            float rs = p0 + p1;
#pragma unroll
            for (int off = 1; off < 16; off <<= 1) rs += __shfl_xor(rs, off, 32);
            lrun[i] = lrun[i] * alpha + rs;
            mrun[i] = mnew;
#pragma unroll
            for (int dt = 0; dt < 4; ++dt) yacc[dt][i] *= alpha;
            Plds[wid][i + 8 * hlf][l16]      = (bf16)p0;
            Plds[wid][i + 8 * hlf][16 + l16] = (bf16)p1;
        }
        __syncthreads();                                   // P visible; K reads done
        v16bf aP = cat(ld8(&Plds[wid][l16][hlf * 16]), ld8(&Plds[wid][l16][hlf * 16 + 8]));
        // P(16x32) x V(32x64) from shared V^T tile
#pragma unroll
        for (int dt = 0; dt < 4; ++dt) {
            v16bf bV = cat(ld8(&Vlds[dt * 16 + l16][hlf * 16]), ld8(&Vlds[dt * 16 + l16][hlf * 16 + 8]));
            yacc[dt] = wmma_bf16(aP, bV, yacc[dt]);
        }
    }

    // normalize and write y as [B,T,C] bf16 for the projection GEMM
#pragma unroll
    for (int dt = 0; dt < 4; ++dt) {
        int c = h * D_ + dt * 16 + l16;
#pragma unroll
        for (int i = 0; i < 8; ++i) {
            int trow = qw + i + 8 * hlf;
            yws[((size_t)b * T_ + trow) * C_ + c] = (bf16)(yacc[dt][i] / lrun[i]);
        }
    }
}

// ---------------- projection GEMM: [8192,1024] x [1024,1024] -> f32 out, 64x64/wave ----------------

__global__ __launch_bounds__(32) void gemm_proj_kernel(
    const unsigned short* __restrict__ ybu, const unsigned short* __restrict__ wTu,
    const float* __restrict__ bias, float* __restrict__ out) {
    const bf16* yb = (const bf16*)ybu;
    const bf16* wT = (const bf16*)wTu;

    const int lane = threadIdx.x & 31;
    const int hlf = lane >> 4, l16 = lane & 15;
    const int NB = C_ / 64;                                // 16
    int nb = blockIdx.x % NB, mb = blockIdx.x / NB;
    int mbase = mb * 64, nbase = nb * 64;

    v8f acc[4][4];
    v8f zero = {0.f, 0.f, 0.f, 0.f, 0.f, 0.f, 0.f, 0.f};
#pragma unroll
    for (int mt = 0; mt < 4; ++mt)
#pragma unroll
        for (int nt = 0; nt < 4; ++nt) acc[mt][nt] = zero;

    const bf16* arow = yb + (size_t)(mbase + l16) * C_;
    const bf16* brow = wT + (size_t)(nbase + l16) * C_;

    v16bf a0[4], b0[4], a1[4], b1[4];
    load_a4(arow, 0, hlf, a0);
    load_b4(brow, 0, hlf, b0);
    for (int kb = 0; kb < C_; kb += 64) {
        load_a4(arow, kb + 32, hlf, a1);
        load_b4(brow, kb + 32, hlf, b1);
        WMMA_TILE16(a0, b0, acc);
        if (kb + 64 < C_) {
            load_a4(arow, kb + 64, hlf, a0);
            load_b4(brow, kb + 64, hlf, b0);
        }
        WMMA_TILE16(a1, b1, acc);
    }

#pragma unroll
    for (int nt = 0; nt < 4; ++nt) {
        int n = nbase + nt * 16 + l16;
        float bv = bias[n];
#pragma unroll
        for (int mt = 0; mt < 4; ++mt) {
#pragma unroll
            for (int i = 0; i < 8; ++i) {
                int row = mbase + mt * 16 + i + 8 * hlf;
                out[(size_t)row * C_ + n] = acc[mt][nt][i] + bv;
            }
        }
    }
}

// ---------------- host launcher ----------------

extern "C" void kernel_launch(void* const* d_in, const int* in_sizes, int n_in,
                              void* d_out, int out_size, void* d_ws, size_t ws_size,
                              hipStream_t stream) {
    (void)in_sizes; (void)n_in; (void)out_size; (void)ws_size;
    const float* x      = (const float*)d_in[0];
    const float* w_qkv  = (const float*)d_in[1];
    const float* b_qkv  = (const float*)d_in[2];
    const float* w_proj = (const float*)d_in[3];
    const float* b_proj = (const float*)d_in[4];
    float* out = (float*)d_out;

    char* w = (char*)d_ws;
    unsigned short* xb     = (unsigned short*)w; w += (size_t)B_ * T_ * C_ * 2;       // x bf16
    unsigned short* wqkvT  = (unsigned short*)w; w += (size_t)3 * C_ * C_ * 2;        // w_qkv^T bf16
    unsigned short* wprojT = (unsigned short*)w; w += (size_t)C_ * C_ * 2;            // w_proj^T bf16
    unsigned short* qws    = (unsigned short*)w; w += (size_t)B_ * T_ * C_ * 2;       // q [B,H,T,D]
    unsigned short* kws    = (unsigned short*)w; w += (size_t)B_ * T_ * C_ * 2;       // k [B,H,T,D]
    unsigned short* vtws   = (unsigned short*)w; w += (size_t)B_ * T_ * C_ * 2;       // v^T [B,H,D,T]
    unsigned short* yws    = (unsigned short*)w;                                      // y [B,T,C]

    int n_x = B_ * T_ * C_;
    cvt_kernel<<<(n_x + 255) / 256, 256, 0, stream>>>(x, xb, n_x);
    cvt_tr_kernel<<<(3 * C_ * C_ + 255) / 256, 256, 0, stream>>>(w_qkv, wqkvT, C_, 3 * C_);
    cvt_tr_kernel<<<(C_ * C_ + 255) / 256, 256, 0, stream>>>(w_proj, wprojT, C_, C_);

    gemm_qkv_kernel<<<(B_ * T_ / 64) * (3 * C_ / 64), 32, 0, stream>>>(xb, wqkvT, b_qkv, qws, kws, vtws);
    attn_kernel<<<B_ * H_ * (T_ / 64), 128, 0, stream>>>(qws, kws, vtws, yws);
    gemm_proj_kernel<<<(B_ * T_ / 64) * (C_ / 64), 32, 0, stream>>>(yws, wprojT, b_proj, out);
}
